// WindowAttention_20693152432504
// MI455X (gfx1250) — matrix operand order repaired and checked by reading, not compiled, not run on hardware
//
#include <hip/hip_runtime.h>
#include <hip/hip_bf16.h>

#define DIMC 512
#define HEADS 16
#define HD 32
#define NT 64
#define NW 64

typedef __attribute__((ext_vector_type(16))) __bf16 v16bf;
typedef __attribute__((ext_vector_type(8)))  float  v8f;

__device__ __forceinline__ unsigned short f2bf(float f) {
  union { float f; unsigned u; } c; c.f = f;
  unsigned u = c.u;
  u += 0x7FFFu + ((u >> 16) & 1u);   // round-to-nearest-even
  return (unsigned short)(u >> 16);
}

__device__ __forceinline__ unsigned pack2bf(float a, float b) {
  return (unsigned)f2bf(a) | ((unsigned)f2bf(b) << 16);
}

__device__ __forceinline__ v8f zero8() {
  v8f z = {0.f, 0.f, 0.f, 0.f, 0.f, 0.f, 0.f, 0.f};
  return z;
}

// A fragment (16x32 bf16) from row-major [row][k] storage, leading dim ld.
// lane<16: K {k0..k0+7, k0+16..k0+23}; lane>=16: +8. Two 16B loads.
__device__ __forceinline__ v16bf ldA(const unsigned short* base, int ld, int row, int k0, int lane) {
  int kb = k0 + ((lane >> 4) << 3);
  const unsigned short* p = base + row * ld + kb;
  union { uint4 q[2]; v16bf v; } u;
  u.q[0] = *(const uint4*)(p);
  u.q[1] = *(const uint4*)(p + 16);
  return u.v;
}

// B fragment (32x16 bf16) from "BT" row-major [n][k] storage, leading dim ld.
// lane<16: column n0+lane, K {k0..k0+7, k0+16..k0+23}; lane>=16: +8. Two 16B loads.
__device__ __forceinline__ v16bf ldB(const unsigned short* base, int ld, int n0, int k0, int lane) {
  int n = n0 + (lane & 15);
  int kb = k0 + ((lane >> 4) << 3);
  const unsigned short* p = base + n * ld + kb;
  union { uint4 q[2]; v16bf v; } u;
  u.q[0] = *(const uint4*)(p);
  u.q[1] = *(const uint4*)(p + 16);
  return u.v;
}

#define WMMA_BF16(a, b, c) \
  __builtin_amdgcn_wmma_f32_16x16x32_bf16(false, (a), false, (b), (short)0, (c), false, false)

// ---------------- prep: transpose+convert weights to bf16, build fused bias ----
__global__ void prep_kernel(const float* __restrict__ qkv_w, const float* __restrict__ proj_w,
                            const float* __restrict__ rpb, const int* __restrict__ rel_idx,
                            unsigned short* __restrict__ qkvT, unsigned short* __restrict__ projT,
                            float* __restrict__ fbias) {
  int i = blockIdx.x * 256 + threadIdx.x;
  const int QT = 1536 * 512;
  const int PT = 512 * 512;
  const int FB = HEADS * NT * NT;
  if (i < QT) {
    int n = i >> 9, k = i & 511;
    qkvT[i] = f2bf(qkv_w[k * 1536 + n]);
  } else if (i < QT + PT) {
    int j = i - QT;
    int n = j >> 9, k = j & 511;
    projT[j] = f2bf(proj_w[k * 512 + n]);
  } else if (i < QT + PT + FB) {
    int j = i - QT - PT;
    int h = j >> 12, mn = j & 4095;
    fbias[j] = rpb[rel_idx[mn] * HEADS + h];
  }
}

// ---------------- fused window attention: one workgroup per window ------------
__global__ __launch_bounds__(256) void win_attn_kernel(
    const float* __restrict__ x, const float* __restrict__ attn_mask,
    const float* __restrict__ qkv_b, const float* __restrict__ proj_b,
    const unsigned short* __restrict__ qkvT, const unsigned short* __restrict__ projT,
    const float* __restrict__ fbias, float* __restrict__ out) {
  __shared__ __align__(16) unsigned short lq[NT * DIMC];        // q  -> later: attn-out (lo)
  __shared__ __align__(16) unsigned short lk[NT * DIMC];        // k
  __shared__ __align__(16) unsigned short lvT[HEADS * HD * NT]; // v transposed [h][d][tok]
  __shared__ __align__(16) unsigned short lx[NT * DIMC];        // x -> later: probs + o-stage

  const int tid = threadIdx.x;
  const int wv = tid >> 5;
  const int lane = tid & 31;
  const int hb = (lane >> 4) << 3;  // row offset of this half-wave in a C/D tile
  const int ln = lane & 15;         // column lane within tile
  const int b = blockIdx.x;
  const int w = b & (NW - 1);

  // ---- phase 1: load x window, convert to bf16 in LDS (packed b64 stores) ----
  {
    const float4* xs = (const float4*)(x + (size_t)b * NT * DIMC);
#pragma unroll 1
    for (int i = tid; i < NT * DIMC / 4; i += 256) {
      float4 v = xs[i];
      uint2 pk;
      pk.x = pack2bf(v.x, v.y);
      pk.y = pack2bf(v.z, v.w);
      *(uint2*)(lx + i * 4) = pk;
    }
  }
  __syncthreads();

  // ---- phase 2: QKV GEMM [64,512] @ [512,1536], four N-tiles per pass ----
#pragma unroll 1
  for (int np = 0; np < 3; ++np) {
    int n0 = (wv * 12 + np * 4) << 4;
    v8f acc[4][4];
#pragma unroll
    for (int mi = 0; mi < 4; ++mi)
#pragma unroll
      for (int c2 = 0; c2 < 4; ++c2) acc[mi][c2] = zero8();
#pragma unroll 1
    for (int ks = 0; ks < 16; ++ks) {
      v16bf b0 = ldB(qkvT, DIMC, n0,      ks << 5, lane);
      v16bf b1 = ldB(qkvT, DIMC, n0 + 16, ks << 5, lane);
      v16bf b2 = ldB(qkvT, DIMC, n0 + 32, ks << 5, lane);
      v16bf b3 = ldB(qkvT, DIMC, n0 + 48, ks << 5, lane);
#pragma unroll
      for (int mi = 0; mi < 4; ++mi) {
        v16bf af = ldA(lx, DIMC, (mi << 4) + ln, ks << 5, lane);
        acc[mi][0] = WMMA_BF16(af, b0, acc[mi][0]);
        acc[mi][1] = WMMA_BF16(af, b1, acc[mi][1]);
        acc[mi][2] = WMMA_BF16(af, b2, acc[mi][2]);
        acc[mi][3] = WMMA_BF16(af, b3, acc[mi][3]);
      }
    }
    // write-back: each 16-col tile lies wholly in q, k, or v -> scalar branch
#pragma unroll
    for (int c2 = 0; c2 < 4; ++c2) {
      int n0c = n0 + (c2 << 4);
      int seg = __builtin_amdgcn_readfirstlane(n0c);
      int n = n0c + ln;
      float bias = qkv_b[n];
      if (seg < 512) {
#pragma unroll
        for (int mi = 0; mi < 4; ++mi)
#pragma unroll
          for (int r = 0; r < 8; ++r)
            lq[((mi << 4) + hb + r) * DIMC + n] = f2bf(acc[mi][c2][r] + bias);
      } else if (seg < 1024) {
#pragma unroll
        for (int mi = 0; mi < 4; ++mi)
#pragma unroll
          for (int r = 0; r < 8; ++r)
            lk[((mi << 4) + hb + r) * DIMC + (n - 512)] = f2bf(acc[mi][c2][r] + bias);
      } else {
        int c = n - 1024;
        int vb = ((c >> 5) * HD + (c & 31)) * NT;
#pragma unroll
        for (int mi = 0; mi < 4; ++mi)
#pragma unroll
          for (int r = 0; r < 8; ++r)
            lvT[vb + (mi << 4) + hb + r] = f2bf(acc[mi][c2][r] + bias);
      }
    }
  }
  __syncthreads();

  // lx region repurposed: per-wave slice of 8192 ushorts:
  //   [0,4096)     probs [64][64] bf16 (per round)
  //   [4096,8192)  staged attn-out, 2 rounds x [64][32] bf16
  unsigned short* slice = lx + wv * 8192;
  const float scale = 0.17677669529663687f;  // 1/sqrt(32)

#pragma unroll 1
  for (int rd = 0; rd < 2; ++rd) {
    int h = wv + (rd << 3);
    // ---- scores: q_h [64,32] @ k_h^T -> [64,64] (single K step) ----
    v8f acc[4][4];
    v16bf bk[4];
#pragma unroll
    for (int ni = 0; ni < 4; ++ni) {
      bk[ni] = ldB(lk + h * HD, DIMC, ni << 4, 0, lane);
#pragma unroll
      for (int mi = 0; mi < 4; ++mi) acc[mi][ni] = zero8();
    }
#pragma unroll
    for (int mi = 0; mi < 4; ++mi) {
      v16bf aq = ldA(lq + h * HD, DIMC, (mi << 4) + ln, 0, lane);
#pragma unroll
      for (int ni = 0; ni < 4; ++ni)
        acc[mi][ni] = WMMA_BF16(aq, bk[ni], acc[mi][ni]);
    }
    // ---- bias + mask + softmax (rows span 16 lanes x 4 tiles) ----
    const float* fb = fbias + h * (NT * NT);
    const float* mk = attn_mask + w * (NT * NT);
#pragma unroll
    for (int mi = 0; mi < 4; ++mi) {
      float rinv[8];
#pragma unroll
      for (int r = 0; r < 8; ++r) {
        int m = (mi << 4) + hb + r;
        float mx = -1e30f;
#pragma unroll
        for (int ni = 0; ni < 4; ++ni) {
          int n = (ni << 4) + ln;
          float v = acc[mi][ni][r] * scale + fb[m * NT + n] + mk[m * NT + n];
          acc[mi][ni][r] = v;
          mx = fmaxf(mx, v);
        }
        mx = fmaxf(mx, __shfl_xor(mx, 1));
        mx = fmaxf(mx, __shfl_xor(mx, 2));
        mx = fmaxf(mx, __shfl_xor(mx, 4));
        mx = fmaxf(mx, __shfl_xor(mx, 8));
        float s = 0.f;
#pragma unroll
        for (int ni = 0; ni < 4; ++ni) {
          float e = __expf(acc[mi][ni][r] - mx);
          acc[mi][ni][r] = e;
          s += e;
        }
        s += __shfl_xor(s, 1);
        s += __shfl_xor(s, 2);
        s += __shfl_xor(s, 4);
        s += __shfl_xor(s, 8);
        rinv[r] = 1.0f / s;
      }
#pragma unroll
      for (int r = 0; r < 8; ++r) {
        int m = (mi << 4) + hb + r;
#pragma unroll
        for (int ni = 0; ni < 4; ++ni) {
          int n = (ni << 4) + ln;
          slice[m * NT + n] = f2bf(acc[mi][ni][r] * rinv[r]);
        }
      }
    }
    // ---- attn @ v : probs [64,64] @ v_h [64,32] ----
    v8f o[4][2];
#pragma unroll
    for (int mi = 0; mi < 4; ++mi) { o[mi][0] = zero8(); o[mi][1] = zero8(); }
#pragma unroll 1
    for (int ks = 0; ks < 2; ++ks) {
      v16bf bv0 = ldB(lvT + h * HD * NT, NT, 0, ks << 5, lane);
      v16bf bv1 = ldB(lvT + h * HD * NT, NT, 16, ks << 5, lane);
#pragma unroll
      for (int mi = 0; mi < 4; ++mi) {
        v16bf ap = ldA(slice, NT, (mi << 4) + ln, ks << 5, lane);
        o[mi][0] = WMMA_BF16(ap, bv0, o[mi][0]);
        o[mi][1] = WMMA_BF16(ap, bv1, o[mi][1]);
      }
    }
    unsigned short* st = slice + 4096 + (rd << 11);
#pragma unroll
    for (int mi = 0; mi < 4; ++mi)
#pragma unroll
      for (int di = 0; di < 2; ++di)
#pragma unroll
        for (int r = 0; r < 8; ++r) {
          int m = (mi << 4) + hb + r;
          int d = (di << 4) + ln;
          st[m * HD + d] = f2bf(o[mi][di][r]);
        }
  }
  __syncthreads();  // everyone done reading lq/lk/lvT + own probs

  // ---- gather staged heads into lo (reusing lq region): [m][h*32+d] ----
  unsigned short* lo = lq;
#pragma unroll 1
  for (int rd = 0; rd < 2; ++rd) {
    int h = wv + (rd << 3);
    const unsigned short* st = lx + wv * 8192 + 4096 + (rd << 11);
#pragma unroll 1
    for (int it = 0; it < 16; ++it) {
      int i = (it * 32 + lane) * 4;             // 4 consecutive d per lane
      int m = i >> 5, d = i & 31;
      *(uint2*)(lo + m * DIMC + h * HD + d) = *(const uint2*)(st + i);
    }
  }
  __syncthreads();

  // ---- proj GEMM: lo [64,512] @ projT^T -> out [64,512] f32 + bias ----
  float* ob = out + (size_t)b * NT * DIMC;
  {
    int n0 = (wv * 4) << 4;
    v8f acc[4][4];
#pragma unroll
    for (int mi = 0; mi < 4; ++mi)
#pragma unroll
      for (int c2 = 0; c2 < 4; ++c2) acc[mi][c2] = zero8();
#pragma unroll 1
    for (int ks = 0; ks < 16; ++ks) {
      v16bf b0 = ldB(projT, DIMC, n0,      ks << 5, lane);
      v16bf b1 = ldB(projT, DIMC, n0 + 16, ks << 5, lane);
      v16bf b2 = ldB(projT, DIMC, n0 + 32, ks << 5, lane);
      v16bf b3 = ldB(projT, DIMC, n0 + 48, ks << 5, lane);
#pragma unroll
      for (int mi = 0; mi < 4; ++mi) {
        v16bf af = ldA(lo, DIMC, (mi << 4) + ln, ks << 5, lane);
        acc[mi][0] = WMMA_BF16(af, b0, acc[mi][0]);
        acc[mi][1] = WMMA_BF16(af, b1, acc[mi][1]);
        acc[mi][2] = WMMA_BF16(af, b2, acc[mi][2]);
        acc[mi][3] = WMMA_BF16(af, b3, acc[mi][3]);
      }
    }
#pragma unroll
    for (int c2 = 0; c2 < 4; ++c2) {
      int n = n0 + (c2 << 4) + ln;
      float pb = proj_b[n];
#pragma unroll
      for (int mi = 0; mi < 4; ++mi)
#pragma unroll
        for (int r = 0; r < 8; ++r) {
          int m = (mi << 4) + hb + r;
          ob[m * DIMC + n] = acc[mi][c2][r] + pb;
        }
    }
  }
}

extern "C" void kernel_launch(void* const* d_in, const int* in_sizes, int n_in,
                              void* d_out, int out_size, void* d_ws, size_t ws_size,
                              hipStream_t stream) {
  const float* x         = (const float*)d_in[0];
  const float* attn_mask = (const float*)d_in[1];
  const float* qkv_w     = (const float*)d_in[2];
  const float* qkv_b     = (const float*)d_in[3];
  const float* proj_w    = (const float*)d_in[4];
  const float* proj_b    = (const float*)d_in[5];
  const float* rpb       = (const float*)d_in[6];
  const int*   rel_idx   = (const int*)d_in[7];

  unsigned short* qkvT  = (unsigned short*)d_ws;               // 1536*512 bf16
  unsigned short* projT = qkvT + 1536 * 512;                   //  512*512 bf16
  float*          fbias = (float*)(projT + 512 * 512);         // 16*64*64 f32
  float*          outp  = (float*)d_out;

  const int total = 1536 * 512 + 512 * 512 + HEADS * NT * NT;
  hipLaunchKernelGGL(prep_kernel, dim3((total + 255) / 256), dim3(256), 0, stream,
                     qkv_w, proj_w, rpb, rel_idx, qkvT, projT, fbias);
  hipLaunchKernelGGL(win_attn_kernel, dim3(2048), dim3(256), 0, stream,
                     x, attn_mask, qkv_b, proj_b, qkvT, projT, fbias, outp);
}
